// MultiRiskGNN_36240934043846
// MI455X (gfx1250) — matrix operand order repaired and checked by reading, not compiled, not run on hardware
//
#include <hip/hip_runtime.h>
#include <hip/hip_bf16.h>
#include <stdint.h>

// ---------------------------------------------------------------------------
// Problem constants (match reference)
// ---------------------------------------------------------------------------
#define NN      50000
#define EE      400000
#define ETOT    450000          // EE + NN self loops
#define UU      1024
#define H1N     4
#define C1N     128
#define C2N     128
#define F1      512             // H1N*C1N
#define BN_EPS  1e-5f
#define NEG_SLOPE 0.2f
#define NEGINF_ENC 0x007FFFFFu  // enc(-inf) for order-preserving uint max
#define LDS_ROW 520             // 512 + 8 halves pad -> 1040B row stride, conflict-free

typedef __attribute__((ext_vector_type(16))) _Float16 v16h;
typedef __attribute__((ext_vector_type(8)))  _Float16 v8h;
typedef __attribute__((ext_vector_type(8)))  float    v8f;

// order-preserving float<->uint for atomic max
__device__ __forceinline__ unsigned fenc(float f) {
  unsigned u = __float_as_uint(f);
  return (u & 0x80000000u) ? ~u : (u | 0x80000000u);
}
__device__ __forceinline__ float fdec(unsigned u) {
  return (u & 0x80000000u) ? __uint_as_float(u & 0x7FFFFFFFu)
                           : __uint_as_float(~u);
}
__device__ __forceinline__ float lrelu(float v) { return v > 0.f ? v : NEG_SLOPE * v; }
__device__ __forceinline__ float eluf(float v)  { return v > 0.f ? v : (__expf(v) - 1.f); }

// ---------------------------------------------------------------------------
// generic fill
// ---------------------------------------------------------------------------
__global__ void k_fill_u32(uint32_t* p, long n, uint32_t v) {
  long i = (long)blockIdx.x * blockDim.x + threadIdx.x;
  long stride = (long)gridDim.x * blockDim.x;
  for (; i < n; i += stride) p[i] = v;
}

// ---------------------------------------------------------------------------
// s1[h] = dot(W1 head h, a1_src[h]);  d1[h] = dot(W1 head h, a1_dst[h])
// ---------------------------------------------------------------------------
__global__ void k_prep_attn1(const float* __restrict__ W1,
                             const float* __restrict__ a1s,
                             const float* __restrict__ a1d,
                             float* __restrict__ s1d1) {
  int t = threadIdx.x;
  if (t < 8) {
    int h = t & 3;
    const float* av = (t < 4) ? a1s : a1d;
    float acc = 0.f;
    for (int c = 0; c < C1N; ++c) acc += W1[h * C1N + c] * av[h * C1N + c];
    s1d1[t < 4 ? h : 4 + h] = acc;
  }
}

// W2 [512,128] f32 -> f16 (row major, same layout)
__global__ void k_w2_to_f16(const float* __restrict__ W2, _Float16* __restrict__ W2h) {
  int i = blockIdx.x * blockDim.x + threadIdx.x;
  if (i < F1 * C2N) W2h[i] = (_Float16)W2[i];
}

// ---------------------------------------------------------------------------
// GAT1 edge passes: thread per (edge, head)
// ---------------------------------------------------------------------------
__global__ void k_gat1_max(const float* __restrict__ x, const int* __restrict__ ei,
                           const float* __restrict__ s1d1, unsigned* __restrict__ m1enc) {
  long t = (long)blockIdx.x * blockDim.x + threadIdx.x;
  if (t >= (long)ETOT * H1N) return;
  long e = t >> 2; int h = (int)(t & 3);
  int s, d;
  if (e < EE) { s = ei[e]; d = ei[EE + e]; } else { s = d = (int)(e - EE); }
  float ev = lrelu(x[s] * s1d1[h] + x[d] * s1d1[4 + h]);
  atomicMax(&m1enc[(size_t)d * H1N + h], fenc(ev));
}

__global__ void k_gat1_sum(const float* __restrict__ x, const int* __restrict__ ei,
                           const float* __restrict__ s1d1,
                           const unsigned* __restrict__ m1enc,
                           float* __restrict__ denom1, float* __restrict__ numer1) {
  long t = (long)blockIdx.x * blockDim.x + threadIdx.x;
  if (t >= (long)ETOT * H1N) return;
  long e = t >> 2; int h = (int)(t & 3);
  int s, d;
  if (e < EE) { s = ei[e]; d = ei[EE + e]; } else { s = d = (int)(e - EE); }
  float ev = lrelu(x[s] * s1d1[h] + x[d] * s1d1[4 + h]);
  float ex = __expf(ev - fdec(m1enc[(size_t)d * H1N + h]));
  atomicAdd(&denom1[(size_t)d * H1N + h], ex);
  atomicAdd(&numer1[(size_t)d * H1N + h], ex * x[s]);
}

// ---------------------------------------------------------------------------
// S1[n,h] = numer/denom ; accumulate per-head sum / sumsq for BN stats
// statsum layout: [0..3]=sum S1, [4..7]=sum S1^2
// ---------------------------------------------------------------------------
__global__ void k_s1_stats(const float* __restrict__ numer1, const float* __restrict__ denom1,
                           float* __restrict__ S1, float* __restrict__ statsum) {
  int n = blockIdx.x * blockDim.x + threadIdx.x;
  float sv[H1N];
  bool valid = (n < NN);
  #pragma unroll
  for (int h = 0; h < H1N; ++h) {
    float v = 0.f;
    if (valid) {
      v = numer1[(size_t)n * H1N + h] / denom1[(size_t)n * H1N + h];
      S1[(size_t)n * H1N + h] = v;
    }
    sv[h] = v;
  }
  // wave32 reduction, lane 0 commits
  #pragma unroll
  for (int h = 0; h < H1N; ++h) {
    float s = sv[h], q = sv[h] * sv[h];
    #pragma unroll
    for (int m = 16; m >= 1; m >>= 1) {
      s += __shfl_xor(s, m, 32);
      q += __shfl_xor(q, m, 32);
    }
    if ((threadIdx.x & 31) == 0) {
      atomicAdd(&statsum[h], s);
      atomicAdd(&statsum[4 + h], q);
    }
  }
}

// cA[j] = W1[j]*rsqrt(W1[j]^2*var_h + eps)*gamma[j] ; cB[j] = -mu_h*cA[j] + beta[j]
__global__ void k_bn_consts(const float* __restrict__ W1, const float* __restrict__ gamma,
                            const float* __restrict__ beta, const float* __restrict__ statsum,
                            float* __restrict__ cA, float* __restrict__ cB) {
  int j = blockIdx.x * blockDim.x + threadIdx.x;
  if (j >= F1) return;
  int h = j >> 7;
  float inv_n = 1.f / (float)NN;
  float mu = statsum[h] * inv_n;
  float var = statsum[4 + h] * inv_n - mu * mu;
  float rs = rsqrtf(W1[j] * W1[j] * var + BN_EPS);
  float a = W1[j] * rs * gamma[j];
  cA[j] = a;
  cB[j] = -mu * a + beta[j];
}

// ---------------------------------------------------------------------------
// WMMA GEMM: h2 = elu(BN(gat1_out)) @ W2      [50000,512] x [512,128]
// Block = 256 threads = 8 waves; block owns 16 rows of M; wave w owns N tile w.
// Phase 1: all 256 threads cooperatively build the 16x512 f16 A-tile in LDS
//          (BN affine + ELU fused, computed ONCE per block instead of per wave).
// Phase 2: each wave streams A fragments from LDS (2x ds_load_b128, bank-
//          conflict free via 1040B row stride) + B fragments from L2-resident
//          W2, feeding v_wmma_f32_16x16x32_f16 with f32 accumulation.
// ---------------------------------------------------------------------------
__global__ void __launch_bounds__(256)
k_gemm_wmma(const float* __restrict__ S1, const float* __restrict__ cA,
            const float* __restrict__ cB, const _Float16* __restrict__ W2h,
            float* __restrict__ h2) {
  __shared__ _Float16 Atile[16][LDS_ROW];

  const int t    = threadIdx.x;
  const int lane = t & 31;
  const int wave = t >> 5;
  const int hi   = lane >> 4;       // lane half (affects A's K split)
  const int mlo  = lane & 15;
  const int m0   = blockIdx.x * 16;
  const int n0   = wave * 16;

  // ---- phase 1: build A tile (16 rows x 512 cols, f16) ----
  {
    const int row = t >> 4;               // 0..15
    const int c0  = (t & 15) * 32;        // 0..480, single head per 32-run
    const float sv = S1[(size_t)(m0 + row) * H1N + (c0 >> 7)];
    #pragma unroll 8
    for (int kk = 0; kk < 32; ++kk) {
      const int k = c0 + kk;
      float v = sv * cA[k] + cB[k];            // BN affine
      v = v > 0.f ? v : (__expf(v) - 1.f);     // ELU
      Atile[row][k] = (_Float16)v;
    }
  }
  __syncthreads();

  // ---- phase 2: WMMA over K = 512 = 16 * 32 ----
  v8f acc = {};
  #pragma unroll 4
  for (int ks = 0; ks < 16; ++ks) {
    const int k0 = ks * 32;
    // ISA §7.12.2 16-bit A 16x32: lane's elements are two contiguous 8-runs
    const v8h alo = *(const v8h*)&Atile[mlo][k0 + 8 * hi];
    const v8h ahi = *(const v8h*)&Atile[mlo][k0 + 16 + 8 * hi];
    v16h a;
    #pragma unroll
    for (int i = 0; i < 8; ++i) { a[i] = alo[i]; a[8 + i] = ahi[i]; }
    // B 32x16: N = lane&15, K = same two 8-runs as A
    v16h b;
    #pragma unroll
    for (int i = 0; i < 8; ++i) {
      b[i]     = W2h[(size_t)(k0 + 8 * hi + i) * C2N + n0 + mlo];
      b[8 + i] = W2h[(size_t)(k0 + 16 + 8 * hi + i) * C2N + n0 + mlo];
    }
    acc = __builtin_amdgcn_wmma_f32_16x16x32_f16(false, a, false, b,
                                                 (short)0, acc, false, false);
  }

  // C/D layout: N = lane&15, M = 8*hi + r
  const int ncol = n0 + mlo;
  #pragma unroll
  for (int r = 0; r < 8; ++r)
    h2[(size_t)(m0 + 8 * hi + r) * C2N + ncol] = acc[r];
}

// ---------------------------------------------------------------------------
// GAT2 attention logits per node: as2/ad2 = h2 . a2_{src,dst}
// ---------------------------------------------------------------------------
__global__ void k_attn2(const float* __restrict__ h2, const float* __restrict__ a2s,
                        const float* __restrict__ a2d,
                        float* __restrict__ as2, float* __restrict__ ad2) {
  int n = blockIdx.x * blockDim.x + threadIdx.x;
  if (n >= NN) return;
  const float* hp = h2 + (size_t)n * C2N;
  float s = 0.f, d = 0.f;
  for (int c = 0; c < C2N; ++c) { float v = hp[c]; s += v * a2s[c]; d += v * a2d[c]; }
  as2[n] = s; ad2[n] = d;
}

__global__ void k_gat2_max(const int* __restrict__ ei, const float* __restrict__ as2,
                           const float* __restrict__ ad2, unsigned* __restrict__ m2enc) {
  long e = (long)blockIdx.x * blockDim.x + threadIdx.x;
  if (e >= ETOT) return;
  int s, d;
  if (e < EE) { s = ei[e]; d = ei[EE + e]; } else { s = d = (int)(e - EE); }
  atomicMax(&m2enc[d], fenc(lrelu(as2[s] + ad2[d])));
}

__global__ void k_gat2_sum(const int* __restrict__ ei, const float* __restrict__ as2,
                           const float* __restrict__ ad2, const unsigned* __restrict__ m2enc,
                           float* __restrict__ denom2, float* __restrict__ wexp) {
  long e = (long)blockIdx.x * blockDim.x + threadIdx.x;
  if (e >= ETOT) return;
  int s, d;
  if (e < EE) { s = ei[e]; d = ei[EE + e]; } else { s = d = (int)(e - EE); }
  float ex = __expf(lrelu(as2[s] + ad2[d]) - fdec(m2enc[d]));
  atomicAdd(&denom2[d], ex);
  wexp[e] = ex;
}

// acc2[dst,:] += w_e * h2[src,:]   (block = 128 channels, grid = edges)
__global__ void k_gat2_scatter(const int* __restrict__ ei, const float* __restrict__ h2,
                               const float* __restrict__ wexp, float* __restrict__ acc2) {
  long e = blockIdx.x;
  int c = threadIdx.x;
  int s, d;
  if (e < EE) { s = ei[e]; d = ei[EE + e]; } else { s = d = (int)(e - EE); }
  atomicAdd(&acc2[(size_t)d * C2N + c], wexp[e] * h2[(size_t)s * C2N + c]);
}

// h3 = elu(acc2/denom2 + b2); pool into user sums, counts, global sum
__global__ void k_node_final(const float* __restrict__ acc2, const float* __restrict__ denom2,
                             const float* __restrict__ b2, const int* __restrict__ uid,
                             float* __restrict__ usum, float* __restrict__ cnt,
                             float* __restrict__ gsum) {
  int n = blockIdx.x, c = threadIdx.x;
  float v = eluf(acc2[(size_t)n * C2N + c] / denom2[n] + b2[c]);
  int u = uid[n];
  atomicAdd(&usum[(size_t)u * C2N + c], v);
  atomicAdd(&gsum[c], v);
  if (c == 0) atomicAdd(&cnt[u], 1.0f);
}

// heads: one wave per user (lane = hidden unit), block 1024 = group head
__global__ void k_heads(const float* __restrict__ usum, const float* __restrict__ cnt,
                        const float* __restrict__ gsum,
                        const float* __restrict__ Wu1, const float* __restrict__ bu1,
                        const float* __restrict__ Wu2, const float* __restrict__ bu2,
                        const float* __restrict__ Wg1, const float* __restrict__ bg1,
                        const float* __restrict__ Wg2, const float* __restrict__ bg2,
                        float* __restrict__ out) {
  int u = blockIdx.x;
  int j = threadIdx.x;                 // hidden unit 0..31
  bool group = (u == UU);
  const float* emb = group ? gsum : (usum + (size_t)u * C2N);
  float scale = group ? (1.f / (float)NN) : (1.f / fmaxf(cnt[u], 1.f));
  const float* Wa = group ? Wg1 : Wu1;
  const float* ba = group ? bg1 : bu1;
  const float* Wb = group ? Wg2 : Wu2;
  const float* bb = group ? bg2 : bu2;

  float h = ba[j];
  for (int c = 0; c < C2N; ++c) h += emb[c] * scale * Wa[c * 32 + j];
  h = fmaxf(h, 0.f) * Wb[j];
  #pragma unroll
  for (int m = 16; m >= 1; m >>= 1) h += __shfl_xor(h, m, 32);
  if (j == 0) out[u] = 1.f / (1.f + __expf(-(h + bb[0])));
}

// ---------------------------------------------------------------------------
// launch
// ---------------------------------------------------------------------------
extern "C" void kernel_launch(void* const* d_in, const int* in_sizes, int n_in,
                              void* d_out, int out_size, void* d_ws, size_t ws_size,
                              hipStream_t stream) {
  const float* x      = (const float*)d_in[0];
  const int*   ei     = (const int*)d_in[1];
  const int*   uid    = (const int*)d_in[2];
  const float* W1     = (const float*)d_in[3];
  const float* a1s    = (const float*)d_in[4];
  const float* a1d    = (const float*)d_in[5];
  const float* gamma  = (const float*)d_in[7];
  const float* beta   = (const float*)d_in[8];
  const float* W2     = (const float*)d_in[9];
  const float* a2s    = (const float*)d_in[10];
  const float* a2d    = (const float*)d_in[11];
  const float* b2     = (const float*)d_in[12];
  const float* Wu1    = (const float*)d_in[13];
  const float* bu1    = (const float*)d_in[14];
  const float* Wu2    = (const float*)d_in[15];
  const float* bu2    = (const float*)d_in[16];
  const float* Wg1    = (const float*)d_in[17];
  const float* bg1    = (const float*)d_in[18];
  const float* Wg2    = (const float*)d_in[19];
  const float* bg2    = (const float*)d_in[20];
  float* out = (float*)d_out;

  char* ws = (char*)d_ws;
  size_t off = 0;
  auto alloc = [&](size_t bytes) -> size_t {
    size_t o = off;
    off = (off + bytes + 255) & ~(size_t)255;
    return o;
  };
  // constants (written before read)
  size_t o_s1d1 = alloc(8 * 4);
  size_t o_cA   = alloc(F1 * 4);
  size_t o_cB   = alloc(F1 * 4);
  // contiguous zero-init block
  size_t zstart = off;
  size_t o_den1 = alloc((size_t)NN * H1N * 4);
  size_t o_num1 = alloc((size_t)NN * H1N * 4);
  size_t o_den2 = alloc((size_t)NN * 4);
  size_t o_acc2 = alloc((size_t)NN * C2N * 4);
  size_t o_usum = alloc((size_t)UU * C2N * 4);
  size_t o_cnt  = alloc((size_t)UU * 4);
  size_t o_gsum = alloc((size_t)C2N * 4);
  size_t o_stat = alloc(8 * 4);
  size_t zend = off;
  // contiguous -inf-encoded block
  size_t nstart = off;
  size_t o_m1   = alloc((size_t)NN * H1N * 4);
  size_t o_m2   = alloc((size_t)NN * 4);
  size_t nend = off;
  // rest
  size_t o_S1   = alloc((size_t)NN * H1N * 4);
  size_t o_W2h  = alloc((size_t)F1 * C2N * 2);
  size_t o_h2   = alloc((size_t)NN * C2N * 4);
  size_t o_as2  = alloc((size_t)NN * 4);
  size_t o_ad2  = alloc((size_t)NN * 4);
  size_t o_wexp = alloc((size_t)ETOT * 4);
  (void)ws_size; (void)n_in; (void)in_sizes; (void)out_size;

  float*    s1d1  = (float*)(ws + o_s1d1);
  float*    cA    = (float*)(ws + o_cA);
  float*    cB    = (float*)(ws + o_cB);
  float*    den1  = (float*)(ws + o_den1);
  float*    num1  = (float*)(ws + o_num1);
  float*    den2  = (float*)(ws + o_den2);
  float*    acc2  = (float*)(ws + o_acc2);
  float*    usum  = (float*)(ws + o_usum);
  float*    cnt   = (float*)(ws + o_cnt);
  float*    gsum  = (float*)(ws + o_gsum);
  float*    stat  = (float*)(ws + o_stat);
  unsigned* m1enc = (unsigned*)(ws + o_m1);
  unsigned* m2enc = (unsigned*)(ws + o_m2);
  float*    S1    = (float*)(ws + o_S1);
  _Float16* W2h   = (_Float16*)(ws + o_W2h);
  float*    h2    = (float*)(ws + o_h2);
  float*    as2   = (float*)(ws + o_as2);
  float*    ad2   = (float*)(ws + o_ad2);
  float*    wexp  = (float*)(ws + o_wexp);

  // init
  long zc = (long)((zend - zstart) / 4);
  long nc = (long)((nend - nstart) / 4);
  k_fill_u32<<<(int)((zc + 255) / 256), 256, 0, stream>>>((uint32_t*)(ws + zstart), zc, 0u);
  k_fill_u32<<<(int)((nc + 255) / 256), 256, 0, stream>>>((uint32_t*)(ws + nstart), nc, NEGINF_ENC);

  // prep
  k_prep_attn1<<<1, 64, 0, stream>>>(W1, a1s, a1d, s1d1);
  k_w2_to_f16<<<(F1 * C2N + 255) / 256, 256, 0, stream>>>(W2, W2h);

  // GAT1
  long t1 = (long)ETOT * H1N;
  k_gat1_max<<<(int)((t1 + 255) / 256), 256, 0, stream>>>(x, ei, s1d1, m1enc);
  k_gat1_sum<<<(int)((t1 + 255) / 256), 256, 0, stream>>>(x, ei, s1d1, m1enc, den1, num1);
  k_s1_stats<<<(NN + 255) / 256, 256, 0, stream>>>(num1, den1, S1, stat);
  k_bn_consts<<<(F1 + 255) / 256, 256, 0, stream>>>(W1, gamma, beta, stat, cA, cB);

  // fused BN+ELU -> WMMA GEMM -> h2
  k_gemm_wmma<<<NN / 16, 256, 0, stream>>>(S1, cA, cB, W2h, h2);

  // GAT2
  k_attn2<<<(NN + 255) / 256, 256, 0, stream>>>(h2, a2s, a2d, as2, ad2);
  k_gat2_max<<<(ETOT + 255) / 256, 256, 0, stream>>>(ei, as2, ad2, m2enc);
  k_gat2_sum<<<(ETOT + 255) / 256, 256, 0, stream>>>(ei, as2, ad2, m2enc, den2, wexp);
  k_gat2_scatter<<<ETOT, 128, 0, stream>>>(ei, h2, wexp, acc2);
  k_node_final<<<NN, 128, 0, stream>>>(acc2, den2, b2, uid, usum, cnt, gsum);

  // heads: 1024 users + 1 group
  k_heads<<<UU + 1, 32, 0, stream>>>(usum, cnt, gsum, Wu1, bu1, Wu2, bu2,
                                     Wg1, bg1, Wg2, bg2, out);
}
